// DenseMotionNetwork_60627758350526
// MI455X (gfx1250) — matrix operand order repaired and checked by reading, not compiled, not run on hardware
//
#include <hip/hip_runtime.h>
#include <hip/hip_bf16.h>

// ---------------------------------------------------------------------------
// DenseMotionNetwork forward for MI455X (gfx1250), v5.
//
// Conv3d layers -> implicit GEMM on v_wmma_f32_16x16x32_f16.
//  * Activations channel-last f16 (D,H,W,C); weights pre-transposed per layer
//    to f16 (Cout, KS^3, Cin). K-order = (tap major, cin minor) for A and B.
//  * Register-blocked: one wave computes 16 couts x 64 voxels (4 v8f accs);
//    the A fragment is loaded once per K-chunk and reused by 4 WMMAs.
//  * Per chunk: all 10 b128 loads issued as one batch, then 4 WMMAs — lets
//    the scheduler overlap the next chunk's loads with current WMMAs.
//  * Branch-free inner loop: out-of-halo taps redirect the B pointer to a
//    zeroed 4KB page once per tap. Cin%32==16 tail is one straight chunk.
//  * Epilogue fuses bias+BN+ReLU, stores 8 contiguous f16 per lane (b128).
//
// d_in order = setup_inputs() dict order, params in JAX pytree order
// (dict keys sorted alphabetically, lists in order):
//   0 feature  1 audio_kp  2 image_kp
//   3 compress_b  4 compress_w
//   5..8 hg.bn.{beta,gamma,mean,var}  9 hg.conv_b  10 hg.conv_w
//   11+6i: hg.dec[i] {b, bn.beta, bn.gamma, bn.mean, bn.var, w}  i=0..4
//   41+6i: hg.enc[i] {b, bn.beta, bn.gamma, bn.mean, bn.var, w}  i=0..4
//   71 mask_b  72 mask_w  73..76 norm.{beta,gamma,mean,var}  77 occ_b 78 occ_w
//
// d_out: [0,196608) deformation (16,64,64,3) f32; [196608,200704) occlusion.
// Workspace: ~103 MB statically laid out (weight scratch reused per layer).
// ---------------------------------------------------------------------------

typedef __attribute__((ext_vector_type(16))) _Float16 v16h;
typedef __attribute__((ext_vector_type(8)))  _Float16 v8h;
typedef __attribute__((ext_vector_type(8)))  float    v8f;
typedef _Float16 f16;

#define DHW_FULL 65536   // 16*64*64
#define BN_EPS_C 1e-5f

// ---------------------------------------------------------------------------
// Zero a small f16 region (the conv zero-page).
// ---------------------------------------------------------------------------
__global__ void zero_kernel(f16* __restrict__ p, int n) {
  int t = blockIdx.x * blockDim.x + threadIdx.x;
  if (t < n) p[t] = (f16)0.f;
}

// ---------------------------------------------------------------------------
// Weight transpose+convert: f32 (Cout, Cin, KS3) -> f16 (Cout, KS3, Cin)
// ---------------------------------------------------------------------------
__global__ void wtx_kernel(const float* __restrict__ w, f16* __restrict__ out,
                           int Cout, int Cin, int KS3) {
  int t = blockIdx.x * blockDim.x + threadIdx.x;
  int total = Cout * Cin * KS3;
  if (t >= total) return;
  int ci  = t % Cin;
  int r   = t / Cin;
  int tap = r % KS3;
  int co  = r / KS3;
  out[t] = (f16)w[((size_t)co * Cin + ci) * KS3 + tap];
}

// ---------------------------------------------------------------------------
// 1x1x1 compress conv (32->4) + BN + ReLU, f32 out channel-last (DHW,4)
// ---------------------------------------------------------------------------
__global__ void compress_kernel(const float* __restrict__ feat,  // (32, DHW)
                                const float* __restrict__ w,
                                const float* __restrict__ b,
                                const float* __restrict__ nbeta,
                                const float* __restrict__ ngamma,
                                const float* __restrict__ nmean,
                                const float* __restrict__ nvar,
                                float* __restrict__ f) {         // (DHW, 4)
  int v = blockIdx.x * blockDim.x + threadIdx.x;
  if (v >= DHW_FULL) return;
  #pragma unroll
  for (int co = 0; co < 4; ++co) {
    float s = b[co];
    #pragma unroll
    for (int ci = 0; ci < 32; ++ci)
      s += w[co * 32 + ci] * feat[ci * DHW_FULL + v];
    float inv = ngamma[co] * rsqrtf(nvar[co] + BN_EPS_C);
    s = (s - nmean[co]) * inv + nbeta[co];
    f[v * 4 + co] = s > 0.f ? s : 0.f;
  }
}

// ---------------------------------------------------------------------------
// Build hourglass input: heatmaps + trilinear grid-sampled deformed features.
// inp channel-last (DHW, 80) f16, channel c = k1*5 + j (j=0 heat, 1..4 feat).
// ---------------------------------------------------------------------------
__global__ void build_inp_kernel(const float* __restrict__ f,   // (DHW, 4)
                                 const float* __restrict__ akp, // (15,3)
                                 const float* __restrict__ ikp, // (15,3)
                                 f16* __restrict__ inp) {
  int t = blockIdx.x * blockDim.x + threadIdx.x;
  if (t >= 16 * DHW_FULL) return;
  int k1 = t >> 16;
  int v  = t & 65535;
  int d = v >> 12, h = (v >> 6) & 63, w = v & 63;
  float gx = 2.f * w / 63.f - 1.f;
  float gy = 2.f * h / 63.f - 1.f;
  float gz = 2.f * d / 15.f - 1.f;
  float mx = gx, my = gy, mz = gz, heat = 0.f;
  if (k1 > 0) {
    int k = k1 - 1;
    float ax = akp[k*3], ay = akp[k*3+1], az = akp[k*3+2];
    float ix = ikp[k*3], iy = ikp[k*3+1], iz = ikp[k*3+2];
    mx = gx - ax + ix; my = gy - ay + iy; mz = gz - az + iz;
    float da = (gx-ax)*(gx-ax) + (gy-ay)*(gy-ay) + (gz-az)*(gz-az);
    float di = (gx-ix)*(gx-ix) + (gy-iy)*(gy-iy) + (gz-iz)*(gz-iz);
    heat = __expf(-50.f * da) - __expf(-50.f * di);   // 0.5/0.01
  }
  float x = ((mx + 1.f) * 64.f - 1.f) * 0.5f;
  float y = ((my + 1.f) * 64.f - 1.f) * 0.5f;
  float z = ((mz + 1.f) * 16.f - 1.f) * 0.5f;
  float x0f = floorf(x), y0f = floorf(y), z0f = floorf(z);
  float wx = x - x0f, wy = y - y0f, wz = z - z0f;
  float c0 = 0.f, c1 = 0.f, c2 = 0.f, c3 = 0.f;
  const float4* f4 = (const float4*)f;
  #pragma unroll
  for (int dz = 0; dz < 2; ++dz)
    #pragma unroll
    for (int dy = 0; dy < 2; ++dy)
      #pragma unroll
      for (int dx = 0; dx < 2; ++dx) {
        int zi = (int)z0f + dz, yi = (int)y0f + dy, xi = (int)x0f + dx;
        if (zi < 0 || zi >= 16 || yi < 0 || yi >= 64 || xi < 0 || xi >= 64) continue;
        float wgt = (dz ? wz : 1.f - wz) * (dy ? wy : 1.f - wy) * (dx ? wx : 1.f - wx);
        float4 fv = f4[(zi << 12) + (yi << 6) + xi];
        c0 += wgt * fv.x; c1 += wgt * fv.y; c2 += wgt * fv.z; c3 += wgt * fv.w;
      }
  size_t base = (size_t)v * 80 + k1 * 5;
  inp[base]     = (f16)heat;
  inp[base + 1] = (f16)c0;
  inp[base + 2] = (f16)c1;
  inp[base + 3] = (f16)c2;
  inp[base + 4] = (f16)c3;
}

// ---------------------------------------------------------------------------
// Implicit-GEMM conv3d via WMMA f16->f32, channel-last activations.
//   src : f16 (D,H,W,Cin)     wtx : f16 (Cout, KS^3, Cin)
// One wave = 16 couts x 64 voxels (4 sub-tiles). K loop: tap outer, Cin
// chunks of 32 inner; A loaded once per chunk, reused by 4 WMMAs.
// All 10 b128 loads of a chunk are issued before its 4 WMMAs.
// Stores into dst with channel stride dstC at offset dstCoff (for concat).
// ---------------------------------------------------------------------------
template <int KS>
__global__ void __launch_bounds__(256) conv3d_wmma(
    const f16*   __restrict__ src,
    const f16*   __restrict__ wtx,
    const f16*   __restrict__ zerop,   // >= 2048 zeroed f16
    const float* __restrict__ bias,
    const float* __restrict__ bn_beta,
    const float* __restrict__ bn_gamma,
    const float* __restrict__ bn_mean,
    const float* __restrict__ bn_var,
    f16*   __restrict__ dst16,
    float* __restrict__ dst32,
    int Cin, int Cout, int D, int H, int W, int pad, int relu_bn,
    int dstC, int dstCoff) {
  const int KS2 = KS * KS, KS3 = KS * KS * KS;
  const int DHW = D * H * W;
  const int posTiles  = DHW  >> 6;     // 64 voxels per wave (DHW % 64 == 0)
  const int coutTiles = Cout >> 4;
  int wave = (blockIdx.x * blockDim.x + threadIdx.x) >> 5;
  if (wave >= posTiles * coutTiles) return;          // wave-uniform exit
  int lane = threadIdx.x & 31;
  int half = lane >> 4;
  int l16  = lane & 15;
  int tileCo  = wave / posTiles;
  int tilePos = wave - tileCo * posTiles;

  // 4 output voxels per lane (B column N = l16 of each 16-voxel sub-tile).
  int n[4], d0[4], h0[4], w0[4];
  #pragma unroll
  for (int s = 0; s < 4; ++s) {
    n[s]  = (tilePos << 6) + (s << 4) + l16;
    d0[s] = n[s] / (H * W);
    int rm = n[s] - d0[s] * H * W;
    h0[s] = rm / W;
    w0[s] = rm - h0[s] * W;
  }

  const f16* wrow = wtx + (size_t)((tileCo << 4) + l16) * KS3 * Cin; // A row M=l16
  const int aoff = half << 3;   // within-chunk A run offset
  const int boff = half << 3;   // within-chunk B run offset
  const int nFull = Cin >> 5;             // full 32-K chunks (uniform)
  const bool tail = (Cin & 31) != 0;      // remainder is always exactly 16
  const v8h vzero = {};

  v8f acc[4] = {};
  for (int tap = 0; tap < KS3; ++tap) {
    int kd = tap / KS2;
    int r2 = tap - kd * KS2;
    int kh = r2 / KS;
    int kw = r2 - kh * KS;
    // One pointer select per sub-tile per tap; chunk loads are unconditional.
    const f16* bb[4];
    #pragma unroll
    for (int s = 0; s < 4; ++s) {
      int zi = d0[s] + kd - pad, yi = h0[s] + kh - pad, xi = w0[s] + kw - pad;
      bool inb = (unsigned)zi < (unsigned)D && (unsigned)yi < (unsigned)H &&
                 (unsigned)xi < (unsigned)W;
      bb[s] = inb ? (src + (size_t)((zi * H + yi) * W + xi) * Cin) : zerop;
    }
    const f16* abase = wrow + (size_t)tap * Cin;
    for (int cc = 0; cc < nFull; ++cc) {
      int cb = cc << 5;
      // ---- load batch: 2 A + 8 B b128 loads, no consumer in between ----
      v8h a_lo = *(const v8h*)(abase + cb + aoff);
      v8h a_hi = *(const v8h*)(abase + cb + 16 + aoff);
      v8h blo[4], bhi[4];
      #pragma unroll
      for (int s = 0; s < 4; ++s) {
        const f16* bp = bb[s] + cb + boff;
        blo[s] = *(const v8h*)bp;
        bhi[s] = *(const v8h*)(bp + 16);
      }
      // ---- compute batch: 4 WMMAs sharing the A fragment ----
      v16h a = __builtin_shufflevector(a_lo, a_hi,
          0,1,2,3,4,5,6,7, 8,9,10,11,12,13,14,15);
      #pragma unroll
      for (int s = 0; s < 4; ++s) {
        v16h b = __builtin_shufflevector(blo[s], bhi[s],
            0,1,2,3,4,5,6,7, 8,9,10,11,12,13,14,15);
        acc[s] = __builtin_amdgcn_wmma_f32_16x16x32_f16(
            false, a, false, b, (short)0, acc[s], false, false);
      }
    }
    if (tail) {
      int cb = nFull << 5;
      // A: only first 16 K valid; upper 8 elements are zero.
      v8h a_lo = *(const v8h*)(abase + cb + aoff);
      v8h blo[4], bhi[4];
      #pragma unroll
      for (int s = 0; s < 4; ++s) {
        // B: elements 8..15 (K>=16) are zero.
        const f16* bp = bb[s] + cb + boff;
        blo[s] = *(const v8h*)bp;
        bhi[s] = vzero;
      }
      v16h a = __builtin_shufflevector(a_lo, vzero,
          0,1,2,3,4,5,6,7, 8,9,10,11,12,13,14,15);
      #pragma unroll
      for (int s = 0; s < 4; ++s) {
        v16h b = __builtin_shufflevector(blo[s], bhi[s],
            0,1,2,3,4,5,6,7, 8,9,10,11,12,13,14,15);
        acc[s] = __builtin_amdgcn_wmma_f32_16x16x32_f16(
            false, a, false, b, (short)0, acc[s], false, false);
      }
    }
  }

  // D layout: row M = r + 8*half (consecutive couts), col N = l16 (voxel n).
  int cbase = (tileCo << 4) + (half << 3);
  float sc[8], sh[8];
  #pragma unroll
  for (int r = 0; r < 8; ++r) {
    int co = cbase + r;
    if (relu_bn) {
      float inv = bn_gamma[co] * rsqrtf(bn_var[co] + BN_EPS_C);
      sc[r] = inv;
      sh[r] = (bias[co] - bn_mean[co]) * inv + bn_beta[co];
    } else {
      sc[r] = 1.f;
      sh[r] = bias[co];
    }
  }
  #pragma unroll
  for (int s = 0; s < 4; ++s) {
    float vals[8];
    #pragma unroll
    for (int r = 0; r < 8; ++r) {
      float v = acc[s][r] * sc[r] + sh[r];
      if (relu_bn) v = v > 0.f ? v : 0.f;
      vals[r] = v;
    }
    if (dst16) {
      v8h o;
      #pragma unroll
      for (int r = 0; r < 8; ++r) o[r] = (f16)vals[r];
      *(v8h*)(dst16 + (size_t)n[s] * dstC + dstCoff + cbase) = o;
    } else {
      float* p = dst32 + (size_t)n[s] * dstC + dstCoff + cbase;
      float4 o0 = { vals[0], vals[1], vals[2], vals[3] };
      float4 o1 = { vals[4], vals[5], vals[6], vals[7] };
      *(float4*)p       = o0;
      *(float4*)(p + 4) = o1;
    }
  }
}

// ---------------------------------------------------------------------------
// avgpool (1,2,2), channel-last f16: (D,2Ho,2Wo,C) -> (D,Ho,Wo,C)
// ---------------------------------------------------------------------------
__global__ void avgpool_kernel(const f16* __restrict__ src, f16* __restrict__ dst,
                               int C, int D, int Ho, int Wo) {
  int t = blockIdx.x * blockDim.x + threadIdx.x;
  int total = D * Ho * Wo * C;
  if (t >= total) return;
  int c  = t % C;  int r = t / C;
  int wo = r % Wo; r /= Wo;
  int ho = r % Ho; int d = r / Ho;
  int Wi = Wo * 2, Hi = Ho * 2;
  size_t base = (size_t)((d * Hi + ho * 2) * Wi + wo * 2);
  float s = (float)src[base * C + c] + (float)src[(base + 1) * C + c] +
            (float)src[(base + Wi) * C + c] + (float)src[(base + Wi + 1) * C + c];
  dst[t] = (f16)(s * 0.25f);
}

// ---------------------------------------------------------------------------
// nearest upsample x2 in H,W, channel-last f16
// ---------------------------------------------------------------------------
__global__ void up_kernel(const f16* __restrict__ src, f16* __restrict__ dst,
                          int C, int D, int Hi, int Wi) {
  int Ho = Hi * 2, Wo = Wi * 2;
  int t = blockIdx.x * blockDim.x + threadIdx.x;
  int total = D * Ho * Wo * C;
  if (t >= total) return;
  int c  = t % C;  int r = t / C;
  int wo = r % Wo; r /= Wo;
  int ho = r % Ho; int d = r / Ho;
  dst[t] = src[(size_t)((d * Hi + (ho >> 1)) * Wi + (wo >> 1)) * C + c];
}

// ---------------------------------------------------------------------------
// channel-last concat copy: dst[v*Cdst + coff + c] = src[v*Csrc + c]
// ---------------------------------------------------------------------------
__global__ void concat_kernel(const f16* __restrict__ src, f16* __restrict__ dst,
                              int V, int Csrc, int Cdst, int coff) {
  int t = blockIdx.x * blockDim.x + threadIdx.x;
  int total = V * Csrc;
  if (t >= total) return;
  int c = t % Csrc;
  int v = t / Csrc;
  dst[(size_t)v * Cdst + coff + c] = src[t];
}

// ---------------------------------------------------------------------------
// softmax over 16 mask channels + weighted sum of sparse motions.
// mask channel-last (DHW,16) f32; out (d,h,w,3) f32
// ---------------------------------------------------------------------------
__global__ void deformation_kernel(const float* __restrict__ mask,
                                   const float* __restrict__ akp,
                                   const float* __restrict__ ikp,
                                   float* __restrict__ out) {
  int v = blockIdx.x * blockDim.x + threadIdx.x;
  if (v >= DHW_FULL) return;
  int d = v >> 12, h = (v >> 6) & 63, w = v & 63;
  float gx = 2.f * w / 63.f - 1.f;
  float gy = 2.f * h / 63.f - 1.f;
  float gz = 2.f * d / 15.f - 1.f;
  float m[16];
  float mx = -3.402823e38f;
  #pragma unroll
  for (int k = 0; k < 16; ++k) { m[k] = mask[(size_t)v * 16 + k]; mx = fmaxf(mx, m[k]); }
  float s = 0.f;
  #pragma unroll
  for (int k = 0; k < 16; ++k) { m[k] = __expf(m[k] - mx); s += m[k]; }
  float inv = 1.f / s;
  float ox = 0.f, oy = 0.f, oz = 0.f;
  #pragma unroll
  for (int k = 0; k < 16; ++k) {
    float p = m[k] * inv;
    float sx = gx, sy = gy, sz = gz;
    if (k > 0) {
      sx += ikp[(k-1)*3]     - akp[(k-1)*3];
      sy += ikp[(k-1)*3 + 1] - akp[(k-1)*3 + 1];
      sz += ikp[(k-1)*3 + 2] - akp[(k-1)*3 + 2];
    }
    ox += p * sx; oy += p * sy; oz += p * sz;
  }
  out[v * 3 + 0] = ox;
  out[v * 3 + 1] = oy;
  out[v * 3 + 2] = oz;
}

// ---------------------------------------------------------------------------
// occlusion conv2d 7x7 (1792->1) + sigmoid. One block per output pixel.
// PRED channel-last (16,64,64,112); pred2 channel cp = c*16 + d.
// ---------------------------------------------------------------------------
__global__ void __launch_bounds__(256) occlusion_kernel(
    const f16*   __restrict__ pred,
    const float* __restrict__ wgt,   // (1792*49)
    const float* __restrict__ bias,  // (1)
    float* __restrict__ out) {       // (4096)
  __shared__ float red[256];
  int pix = blockIdx.x;
  int h = pix >> 6, w = pix & 63;
  float s = 0.f;
  for (int k = threadIdx.x; k < 1792 * 49; k += 256) {
    int cp = k / 49;
    int r  = k - cp * 49;
    int kh = r / 7, kw = r - kh * 7;
    int c  = cp >> 4, dd = cp & 15;
    int yi = h + kh - 3, xi = w + kw - 3;
    if ((unsigned)yi < 64u && (unsigned)xi < 64u)
      s += wgt[k] * (float)pred[(size_t)((dd << 12) + (yi << 6) + xi) * 112 + c];
  }
  red[threadIdx.x] = s;
  __syncthreads();
  for (int off = 128; off > 0; off >>= 1) {
    if (threadIdx.x < off) red[threadIdx.x] += red[threadIdx.x + off];
    __syncthreads();
  }
  if (threadIdx.x == 0) {
    float v = red[0] + bias[0];
    out[pix] = 1.f / (1.f + __expf(-v));
  }
}

// ---------------------------------------------------------------------------
// Host orchestration
// ---------------------------------------------------------------------------
static inline int ceil_div(int a, int b) { return (a + b - 1) / b; }

extern "C" void kernel_launch(void* const* d_in, const int* in_sizes, int n_in,
                              void* d_out, int out_size, void* d_ws, size_t ws_size,
                              hipStream_t stream) {
  (void)in_sizes; (void)n_in; (void)out_size; (void)ws_size;

  const float* feature    = (const float*)d_in[0];
  const float* akp        = (const float*)d_in[1];
  const float* ikp        = (const float*)d_in[2];
  const float* compress_b = (const float*)d_in[3];
  const float* compress_w = (const float*)d_in[4];
  const float* hg_beta    = (const float*)d_in[5];
  const float* hg_gamma   = (const float*)d_in[6];
  const float* hg_mean    = (const float*)d_in[7];
  const float* hg_var     = (const float*)d_in[8];
  const float* hg_conv_b  = (const float*)d_in[9];
  const float* hg_conv_w  = (const float*)d_in[10];
  const float* mask_b     = (const float*)d_in[71];
  const float* mask_w     = (const float*)d_in[72];
  const float* norm_beta  = (const float*)d_in[73];
  const float* norm_gamma = (const float*)d_in[74];
  const float* norm_mean  = (const float*)d_in[75];
  const float* norm_var   = (const float*)d_in[76];
  const float* occ_b      = (const float*)d_in[77];
  const float* occ_w      = (const float*)d_in[78];

  struct Blk { const float *b, *beta, *gamma, *mean, *var, *w; };
  Blk dec[5], enc[5];
  for (int i = 0; i < 5; ++i) {
    int bs = 11 + 6 * i;
    dec[i] = { (const float*)d_in[bs], (const float*)d_in[bs+1], (const float*)d_in[bs+2],
               (const float*)d_in[bs+3], (const float*)d_in[bs+4], (const float*)d_in[bs+5] };
    int be = 41 + 6 * i;
    enc[i] = { (const float*)d_in[be], (const float*)d_in[be+1], (const float*)d_in[be+2],
               (const float*)d_in[be+3], (const float*)d_in[be+4], (const float*)d_in[be+5] };
  }

  // Workspace layout (bytes) — requires ws_size >= ~103 MB. 64B-aligned blocks.
  char* ws = (char*)d_ws;
  const size_t OFF_F    = 0;           // f32 (DHW,4)           1,048,576
  const size_t OFF_MASK = 1048576;     // f32 (DHW,16)          4,194,304
  const size_t OFF_X0   = 5242880;     // f16 (DHW,80)         10,485,760
  const size_t OFF_T    = 15728640;    // f16 pre-pool max      8,388,608
  const size_t OFF_SK0  = 24117248;    // f16 (16*32*32,64)     2,097,152
  const size_t OFF_SK1  = 26214400;    // f16 (16*16*16,128)    1,048,576
  const size_t OFF_SK2  = 27262976;    // f16 (16*8*8,256)        524,288
  const size_t OFF_SK3  = 27787264;    // f16 (16*4*4,512)        262,144
  const size_t OFF_BOT  = 28049408;    // f16 (16*2*2,1024)       131,072
  const size_t OFF_U    = 28180480;    // f16 upsample max     16,777,216
  const size_t OFF_CAT  = 44957696;    // f16 concat max       14,680,064
  const size_t OFF_PRED = 59637760;    // f16 (DHW,112)        14,680,064
  const size_t OFF_WTX  = 74317824;    // f16 weight scratch   28,311,552
  const size_t OFF_ZP   = 102629376;   // f16 zero page             4,096

  float* F     = (float*)(ws + OFF_F);
  float* MASKL = (float*)(ws + OFF_MASK);
  f16* X0   = (f16*)(ws + OFF_X0);
  f16* TB   = (f16*)(ws + OFF_T);
  f16* SK[4] = { (f16*)(ws + OFF_SK0), (f16*)(ws + OFF_SK1),
                 (f16*)(ws + OFF_SK2), (f16*)(ws + OFF_SK3) };
  f16* BOT  = (f16*)(ws + OFF_BOT);
  f16* UB   = (f16*)(ws + OFF_U);
  f16* CAT  = (f16*)(ws + OFF_CAT);
  f16* PRED = (f16*)(ws + OFF_PRED);
  f16* WTX  = (f16*)(ws + OFF_WTX);
  f16* ZP   = (f16*)(ws + OFF_ZP);

  // 0) zero page for out-of-halo loads
  zero_kernel<<<8, 256, 0, stream>>>(ZP, 2048);

  // 1) compress + BN + ReLU -> F (DHW,4)
  compress_kernel<<<ceil_div(DHW_FULL, 256), 256, 0, stream>>>(
      feature, compress_w, compress_b, norm_beta, norm_gamma, norm_mean, norm_var, F);

  // 2) heatmaps + deformed features -> X0 (DHW,80) f16
  build_inp_kernel<<<ceil_div(16 * DHW_FULL, 256), 256, 0, stream>>>(F, akp, ikp, X0);

  // 3) encoder
  const int encCout[5] = { 64, 128, 256, 512, 1024 };
  const f16* cur = X0;
  int cinC = 80, H = 64, W = 64;
  for (int i = 0; i < 5; ++i) {
    int Cout = encCout[i];
    wtx_kernel<<<ceil_div(Cout * cinC * 27, 256), 256, 0, stream>>>(
        enc[i].w, WTX, Cout, cinC, 27);
    int waves = (Cout >> 4) * ((16 * H * W) >> 6);
    conv3d_wmma<3><<<ceil_div(waves, 8), 256, 0, stream>>>(
        cur, WTX, ZP, enc[i].b, enc[i].beta, enc[i].gamma, enc[i].mean, enc[i].var,
        TB, nullptr, cinC, Cout, 16, H, W, 1, 1, Cout, 0);
    f16* pooled = (i < 4) ? SK[i] : BOT;
    int Ho = H / 2, Wo = W / 2;
    avgpool_kernel<<<ceil_div(Cout * 16 * Ho * Wo, 256), 256, 0, stream>>>(
        TB, pooled, Cout, 16, Ho, Wo);
    cur = pooled; cinC = Cout; H = Ho; W = Wo;
  }

  // 4) decoder (cur = BOT 1024 @ 16x2x2)
  const int decCout[5]  = { 512, 256, 128, 64, 32 };
  const int skipC[5]    = { 512, 256, 128, 64, 80 };
  const f16* skipP[5]   = { SK[3], SK[2], SK[1], SK[0], X0 };
  for (int i = 0; i < 5; ++i) {
    up_kernel<<<ceil_div(cinC * 16 * H * 2 * W * 2, 256), 256, 0, stream>>>(
        cur, UB, cinC, 16, H, W);
    H *= 2; W *= 2;
    int Cout = decCout[i];
    int Cdst = Cout + skipC[i];
    int V = 16 * H * W;
    wtx_kernel<<<ceil_div(Cout * cinC * 27, 256), 256, 0, stream>>>(
        dec[i].w, WTX, Cout, cinC, 27);
    int waves = (Cout >> 4) * (V >> 6);
    conv3d_wmma<3><<<ceil_div(waves, 8), 256, 0, stream>>>(
        UB, WTX, ZP, dec[i].b, dec[i].beta, dec[i].gamma, dec[i].mean, dec[i].var,
        CAT, nullptr, cinC, Cout, 16, H, W, 1, 1, Cdst, 0);
    concat_kernel<<<ceil_div(V * skipC[i], 256), 256, 0, stream>>>(
        skipP[i], CAT, V, skipC[i], Cdst, Cout);
    cur = CAT; cinC = Cdst;
  }

  // 5) final conv 112->112 + BN + ReLU  (H=W=64, cinC=112)
  {
    wtx_kernel<<<ceil_div(112 * 112 * 27, 256), 256, 0, stream>>>(
        hg_conv_w, WTX, 112, 112, 27);
    int waves = (112 >> 4) * (DHW_FULL >> 6);
    conv3d_wmma<3><<<ceil_div(waves, 8), 256, 0, stream>>>(
        cur, WTX, ZP, hg_conv_b, hg_beta, hg_gamma, hg_mean, hg_var,
        PRED, nullptr, 112, 112, 16, 64, 64, 1, 1, 112, 0);
  }

  // 6) mask conv 7x7x7 (112->16) -> MASKL (DHW,16) f32 logits
  {
    wtx_kernel<<<ceil_div(16 * 112 * 343, 256), 256, 0, stream>>>(
        mask_w, WTX, 16, 112, 343);
    int waves = (16 >> 4) * (DHW_FULL >> 6);
    conv3d_wmma<7><<<ceil_div(waves, 8), 256, 0, stream>>>(
        PRED, WTX, ZP, mask_b, nullptr, nullptr, nullptr, nullptr,
        nullptr, MASKL, 112, 16, 16, 64, 64, 3, 0, 16, 0);
  }

  // 7) softmax + deformation -> d_out[0 .. 196607]
  deformation_kernel<<<ceil_div(DHW_FULL, 256), 256, 0, stream>>>(
      MASKL, akp, ikp, (float*)d_out);

  // 8) occlusion conv2d 7x7 + sigmoid -> d_out[196608 .. 200703]
  occlusion_kernel<<<4096, 256, 0, stream>>>(
      PRED, occ_w, occ_b, (float*)d_out + 196608);
}
